// MultiHeadAttention_33612414059064
// MI455X (gfx1250) — matrix operand order repaired and checked by reading, not compiled, not run on hardware
//
#include <hip/hip_runtime.h>
#include <hip/hip_bf16.h>

// ---------------- problem dims ----------------
#define B_  4
#define T_  2048
#define D_  1024
#define H_  16
#define QK_ 64
#define M_  (B_*T_)   // 8192 rows of activations

typedef __attribute__((ext_vector_type(16))) __bf16 v16bf;
typedef __attribute__((ext_vector_type(8)))  __bf16 v8bf;
typedef __attribute__((ext_vector_type(8)))  float  v8f;

static __device__ __forceinline__ v8f zero8() {
  v8f z;
#pragma unroll
  for (int i = 0; i < 8; ++i) z[i] = 0.0f;
  return z;
}

// A-operand fragment (16x32 bf16) from a row-major matrix.
// ISA 7.12.2: row m = lane%16; element j -> k = k0 + (j/8)*16 + (lane/16)*8 + j%8
static __device__ __forceinline__ v16bf load_fragA(const __bf16* base, int lane, int k0, int ld) {
  const int m = lane & 15;
  const __bf16* p = base + (size_t)m * ld + k0 + ((lane >> 4) << 3);
  v8bf lo = *reinterpret_cast<const v8bf*>(p);
  v8bf hi = *reinterpret_cast<const v8bf*>(p + 16);
  v16bf r;
#pragma unroll
  for (int j = 0; j < 8; ++j) { r[j] = lo[j]; r[j + 8] = hi[j]; }
  return r;
}

// B-operand fragment (32x16 bf16) from an N-major (transposed) matrix.
// n = lane%16; element j -> k = k0 + (j/8)*16 + (lane/16)*8 + j%8 (same as A).
static __device__ __forceinline__ v16bf load_fragB(const __bf16* base, int lane, int k0, int ld) {
  const int n = lane & 15;
  const __bf16* p = base + (size_t)n * ld + k0 + ((lane >> 4) << 3);
  v8bf lo = *reinterpret_cast<const v8bf*>(p);
  v8bf hi = *reinterpret_cast<const v8bf*>(p + 16);
  v16bf r;
#pragma unroll
  for (int j = 0; j < 8; ++j) { r[j] = lo[j]; r[j + 8] = hi[j]; }
  return r;
}

static __device__ __forceinline__ v8f wmma_bf16(v16bf a, v16bf b, v8f c) {
  return __builtin_amdgcn_wmma_f32_16x16x32_bf16(false, a, false, b, (short)0, c, false, false);
}

// Reductions within each 16-lane half of a wave32 (rows live per-half in WMMA C layout).
static __device__ __forceinline__ float hredadd(float v) {
  v += __shfl_xor(v, 1, 32);
  v += __shfl_xor(v, 2, 32);
  v += __shfl_xor(v, 4, 32);
  v += __shfl_xor(v, 8, 32);
  return v;
}
static __device__ __forceinline__ float hredmax(float v) {
  v = fmaxf(v, __shfl_xor(v, 1, 32));
  v = fmaxf(v, __shfl_xor(v, 2, 32));
  v = fmaxf(v, __shfl_xor(v, 4, 32));
  v = fmaxf(v, __shfl_xor(v, 8, 32));
  return v;
}

// ---------------- conversion kernels ----------------
__global__ void f32_to_bf16_kernel(const float* __restrict__ in, __bf16* __restrict__ out, int n) {
  int i = blockIdx.x * blockDim.x + threadIdx.x;
  if (i < n) out[i] = (__bf16)in[i];
}

// out[n][k] = (bf16) in[k][n]   (in is K x N row-major)
__global__ void transpose_f32_to_bf16(const float* __restrict__ in, __bf16* __restrict__ out,
                                      int K, int N) {
  __shared__ float tile[16][17];
  const int nb = blockIdx.x * 16, kb = blockIdx.y * 16;
  const int tx = threadIdx.x, ty = threadIdx.y;
  tile[ty][tx] = in[(size_t)(kb + ty) * N + nb + tx];
  __syncthreads();
  out[(size_t)(nb + ty) * K + kb + tx] = (__bf16)tile[tx][ty];
}

// ---------------- fused QKV projection (+bias, +RMSnorm, +RoPE / V-transpose) ----------------
// One wave computes a 32-row x 64-col (one head) tile. K-loop is unrolled x2 with ping-pong
// fragment sets: no register rotation, each 8-WMMA group overlaps the other set's 12 loads.
__global__ __launch_bounds__(32)
void gemm_qkv_proj(const __bf16* __restrict__ X, const __bf16* __restrict__ Wt,
                   const float* __restrict__ bias, const float* __restrict__ scale64,
                   const int* __restrict__ positions,
                   __bf16* __restrict__ outR, __bf16* __restrict__ outVT, int mode) {
  const int lane = threadIdx.x;
  const int col = lane & 15, half = lane >> 4;
  const int h = blockIdx.x;           // head index (64-wide column group)
  const int n0 = h * 64;
  const int m0 = blockIdx.y * 32;

  v8f acc[2][4];
#pragma unroll
  for (int m = 0; m < 2; ++m)
#pragma unroll
    for (int t = 0; t < 4; ++t) acc[m][t] = zero8();

  const __bf16* Xr[2] = { X + (size_t)m0 * D_, X + (size_t)(m0 + 16) * D_ };

  v16bf a0[2], b0[4], a1[2], b1[4];
#pragma unroll
  for (int m = 0; m < 2; ++m) a0[m] = load_fragA(Xr[m], lane, 0, D_);
#pragma unroll
  for (int t = 0; t < 4; ++t) b0[t] = load_fragB(Wt + (size_t)(n0 + t * 16) * D_, lane, 0, D_);

  for (int k0 = 0; k0 < D_; k0 += 64) {   // D_/32 = 32 steps -> 16 even iterations
    // fill set 1 @ k0+32 while consuming set 0
#pragma unroll
    for (int m = 0; m < 2; ++m) a1[m] = load_fragA(Xr[m], lane, k0 + 32, D_);
#pragma unroll
    for (int t = 0; t < 4; ++t) b1[t] = load_fragB(Wt + (size_t)(n0 + t * 16) * D_, lane, k0 + 32, D_);
#pragma unroll
    for (int m = 0; m < 2; ++m)
#pragma unroll
      for (int t = 0; t < 4; ++t) acc[m][t] = wmma_bf16(a0[m], b0[t], acc[m][t]);

    // fill set 0 @ k0+64 while consuming set 1
    if (k0 + 64 < D_) {
#pragma unroll
      for (int m = 0; m < 2; ++m) a0[m] = load_fragA(Xr[m], lane, k0 + 64, D_);
#pragma unroll
      for (int t = 0; t < 4; ++t) b0[t] = load_fragB(Wt + (size_t)(n0 + t * 16) * D_, lane, k0 + 64, D_);
    }
#pragma unroll
    for (int m = 0; m < 2; ++m)
#pragma unroll
      for (int t = 0; t < 4; ++t) acc[m][t] = wmma_bf16(a1[m], b1[t], acc[m][t]);
  }

#pragma unroll
  for (int m = 0; m < 2; ++m) {
    float f[4][8];
#pragma unroll
    for (int t = 0; t < 4; ++t)
#pragma unroll
      for (int r = 0; r < 8; ++r)
        f[t][r] = acc[m][t][r] + bias[n0 + t * 16 + col];

    if (mode == 0) {
      // RMS norm over the 64-wide head dim (4 col groups x 16 lanes per row)
      float rstd[8];
#pragma unroll
      for (int r = 0; r < 8; ++r) {
        float ss = f[0][r] * f[0][r] + f[1][r] * f[1][r] + f[2][r] * f[2][r] + f[3][r] * f[3][r];
        ss = hredadd(ss);
        rstd[r] = rsqrtf(ss * (1.0f / 64.0f) + 1e-6f);
      }
#pragma unroll
      for (int t = 0; t < 4; ++t) {
        const float g = 1.0f + scale64[t * 16 + col];
#pragma unroll
        for (int r = 0; r < 8; ++r) f[t][r] *= rstd[r] * g;
      }
      // RoPE: pairs (i, i+32); i = t*16+col for t in {0,1}. timescale = 10000^(i/32)
#pragma unroll
      for (int t = 0; t < 2; ++t) {
        const float invts = exp2f(-(float)(t * 16 + col) * (13.28771237954945f / 32.0f));
#pragma unroll
        for (int r = 0; r < 8; ++r) {
          const int row = m0 + m * 16 + r + 8 * half;
          const float ang = (float)positions[row] * invts;
          const float s = __sinf(ang), c = __cosf(ang);
          const float x1 = f[t][r], x2 = f[t + 2][r];
          f[t][r]     = x1 * c - x2 * s;
          f[t + 2][r] = x2 * c + x1 * s;
        }
      }
#pragma unroll
      for (int t = 0; t < 4; ++t)
#pragma unroll
        for (int r = 0; r < 8; ++r)
          outR[(size_t)(m0 + m * 16 + r + 8 * half) * D_ + n0 + t * 16 + col] = (__bf16)f[t][r];
    } else {
      // V path: store transposed [b][h][d][T] so P*V B-fragments load contiguously over keys
#pragma unroll
      for (int t = 0; t < 4; ++t)
#pragma unroll
        for (int r = 0; r < 8; ++r) {
          const int row = m0 + m * 16 + r + 8 * half;
          const int bb = row >> 11;           // /T_
          const int tq = row & (T_ - 1);
          const int d  = t * 16 + col;
          outVT[(((size_t)bb * H_ + h) * QK_ + d) * T_ + tq] = (__bf16)f[t][r];
        }
    }
  }
}

// ---------------- flash attention (per b, h, 32-row q tile) ----------------
// Next-chunk K fragments and current-chunk V fragments are issued right after the logit
// WMMAs so they overlap the online-softmax VALU work.
__global__ __launch_bounds__(32)
void flash_attn_fwd(const __bf16* __restrict__ Q, const __bf16* __restrict__ K,
                    const __bf16* __restrict__ VT, const float* __restrict__ head_scale,
                    __bf16* __restrict__ O) {
  __shared__ __bf16 P[32][32];   // probs tile, C-layout -> A-layout relay through LDS
  const int lane = threadIdx.x;
  const int col = lane & 15, half = lane >> 4;
  const int qt = blockIdx.x, h = blockIdx.y, b = blockIdx.z;
  const int rbase = b * T_ + qt * 32;

  v16bf aq[2][2];
#pragma unroll
  for (int m = 0; m < 2; ++m) {
    const __bf16* Qb = Q + (size_t)(rbase + m * 16) * D_ + h * QK_;
    aq[m][0] = load_fragA(Qb, lane, 0, D_);
    aq[m][1] = load_fragA(Qb, lane, 32, D_);
  }

  const __bf16* Kb = K + (size_t)b * T_ * D_ + h * QK_;
  const __bf16* Vb = VT + ((size_t)b * H_ + h) * QK_ * T_;

  v8f o[2][4];
#pragma unroll
  for (int m = 0; m < 2; ++m)
#pragma unroll
    for (int v = 0; v < 4; ++v) o[m][v] = zero8();
  float mrun[2][8], lrun[2][8];
#pragma unroll
  for (int m = 0; m < 2; ++m)
#pragma unroll
    for (int r = 0; r < 8; ++r) { mrun[m][r] = -3.0e38f; lrun[m][r] = 0.0f; }

  const int kend = qt * 32 + 32;   // causal: keys <= last q row of tile

  v16bf kf[2][2], kn[2][2];        // [key-half][d-half]
#pragma unroll
  for (int t = 0; t < 2; ++t) {
    kf[t][0] = load_fragB(Kb + (size_t)(t * 16) * D_, lane, 0, D_);
    kf[t][1] = load_fragB(Kb + (size_t)(t * 16) * D_, lane, 32, D_);
  }

  for (int kc = 0; kc < kend; kc += 32) {
    // logits = Q . K^T / sqrt(64) for keys [kc, kc+32)
    v8f c[2][2];
#pragma unroll
    for (int m = 0; m < 2; ++m)
#pragma unroll
      for (int t = 0; t < 2; ++t) {
        c[m][t] = zero8();
        c[m][t] = wmma_bf16(aq[m][0], kf[t][0], c[m][t]);
        c[m][t] = wmma_bf16(aq[m][1], kf[t][1], c[m][t]);
      }

    // issue next-chunk K loads + current-chunk V loads early (overlap softmax VALU)
    const int kcn = kc + 32;
    if (kcn < kend) {
#pragma unroll
      for (int t = 0; t < 2; ++t) {
        kn[t][0] = load_fragB(Kb + (size_t)(kcn + t * 16) * D_, lane, 0, D_);
        kn[t][1] = load_fragB(Kb + (size_t)(kcn + t * 16) * D_, lane, 32, D_);
      }
    }
    v16bf vf[4];
#pragma unroll
    for (int v = 0; v < 4; ++v)
      vf[v] = load_fragB(Vb + (size_t)(v * 16) * T_ + kc, lane, 0, T_);

    // online softmax (rows live in 16-lane halves)
    float pl[2][2][8];
#pragma unroll
    for (int m = 0; m < 2; ++m) {
#pragma unroll
      for (int r = 0; r < 8; ++r) {
        const int qr = qt * 32 + m * 16 + r + 8 * half;
        const int kk = kc + col;
        pl[m][0][r] = (kk      <= qr) ? c[m][0][r] * 0.125f : -3.0e38f;
        pl[m][1][r] = (kk + 16 <= qr) ? c[m][1][r] * 0.125f : -3.0e38f;
      }
#pragma unroll
      for (int r = 0; r < 8; ++r) {
        const float mx   = hredmax(fmaxf(pl[m][0][r], pl[m][1][r]));
        const float mnew = fmaxf(mrun[m][r], mx);
        const float p0 = __expf(pl[m][0][r] - mnew);
        const float p1 = __expf(pl[m][1][r] - mnew);
        pl[m][0][r] = p0; pl[m][1][r] = p1;
        const float alpha = __expf(mrun[m][r] - mnew);
        lrun[m][r] = lrun[m][r] * alpha + hredadd(p0 + p1);
        mrun[m][r] = mnew;
#pragma unroll
        for (int v = 0; v < 4; ++v) o[m][v][r] *= alpha;
      }
    }

    // C-layout -> A-layout relay through LDS
#pragma unroll
    for (int m = 0; m < 2; ++m)
#pragma unroll
      for (int t = 0; t < 2; ++t)
#pragma unroll
        for (int r = 0; r < 8; ++r)
          P[m * 16 + r + 8 * half][t * 16 + col] = (__bf16)pl[m][t][r];
    __syncthreads();
#pragma unroll
    for (int m = 0; m < 2; ++m) {
      const v16bf pf = load_fragA(&P[m * 16][0], lane, 0, 32);
#pragma unroll
      for (int v = 0; v < 4; ++v) o[m][v] = wmma_bf16(pf, vf[v], o[m][v]);
    }
    __syncthreads();

    if (kcn < kend) {
#pragma unroll
      for (int t = 0; t < 2; ++t) { kf[t][0] = kn[t][0]; kf[t][1] = kn[t][1]; }
    }
  }

  const float hs = 1.0f + head_scale[h];
#pragma unroll
  for (int m = 0; m < 2; ++m)
#pragma unroll
    for (int r = 0; r < 8; ++r) {
      const float inv = hs / lrun[m][r];
#pragma unroll
      for (int v = 0; v < 4; ++v)
        O[(size_t)(rbase + m * 16 + r + 8 * half) * D_ + h * QK_ + v * 16 + col] =
            (__bf16)(o[m][v][r] * inv);
    }
}

// ---------------- output projection (32x64 tile per wave, ping-pong unrolled x2) ----------------
__global__ __launch_bounds__(32)
void gemm_out_proj(const __bf16* __restrict__ A, const __bf16* __restrict__ WoT,
                   const float* __restrict__ bias, float* __restrict__ out) {
  const int lane = threadIdx.x;
  const int col = lane & 15, half = lane >> 4;
  const int n0 = blockIdx.x * 64;
  const int m0 = blockIdx.y * 32;

  v8f acc[2][4];
#pragma unroll
  for (int m = 0; m < 2; ++m)
#pragma unroll
    for (int t = 0; t < 4; ++t) acc[m][t] = zero8();

  const __bf16* Ar[2] = { A + (size_t)m0 * D_, A + (size_t)(m0 + 16) * D_ };

  v16bf a0[2], b0[4], a1[2], b1[4];
#pragma unroll
  for (int m = 0; m < 2; ++m) a0[m] = load_fragA(Ar[m], lane, 0, D_);
#pragma unroll
  for (int t = 0; t < 4; ++t) b0[t] = load_fragB(WoT + (size_t)(n0 + t * 16) * D_, lane, 0, D_);

  for (int k0 = 0; k0 < D_; k0 += 64) {
#pragma unroll
    for (int m = 0; m < 2; ++m) a1[m] = load_fragA(Ar[m], lane, k0 + 32, D_);
#pragma unroll
    for (int t = 0; t < 4; ++t) b1[t] = load_fragB(WoT + (size_t)(n0 + t * 16) * D_, lane, k0 + 32, D_);
#pragma unroll
    for (int m = 0; m < 2; ++m)
#pragma unroll
      for (int t = 0; t < 4; ++t) acc[m][t] = wmma_bf16(a0[m], b0[t], acc[m][t]);

    if (k0 + 64 < D_) {
#pragma unroll
      for (int m = 0; m < 2; ++m) a0[m] = load_fragA(Ar[m], lane, k0 + 64, D_);
#pragma unroll
      for (int t = 0; t < 4; ++t) b0[t] = load_fragB(WoT + (size_t)(n0 + t * 16) * D_, lane, k0 + 64, D_);
    }
#pragma unroll
    for (int m = 0; m < 2; ++m)
#pragma unroll
      for (int t = 0; t < 4; ++t) acc[m][t] = wmma_bf16(a1[m], b1[t], acc[m][t]);
  }

#pragma unroll
  for (int m = 0; m < 2; ++m)
#pragma unroll
    for (int t = 0; t < 4; ++t)
#pragma unroll
      for (int r = 0; r < 8; ++r)
        out[(size_t)(m0 + m * 16 + r + 8 * half) * D_ + n0 + t * 16 + col] =
            acc[m][t][r] + bias[n0 + t * 16 + col];
}

// ---------------- launch ----------------
extern "C" void kernel_launch(void* const* d_in, const int* in_sizes, int n_in,
                              void* d_out, int out_size, void* d_ws, size_t ws_size,
                              hipStream_t stream) {
  (void)in_sizes; (void)n_in; (void)out_size; (void)ws_size;
  const float* q    = (const float*)d_in[0];
  const float* kv   = (const float*)d_in[1];
  /* d_in[2] = mask: causal, reconstructed analytically */
  const int*   qpos = (const int*)d_in[3];
  const int*   kpos = (const int*)d_in[4];
  const float* Wq   = (const float*)d_in[5];
  const float* bq   = (const float*)d_in[6];
  const float* Wk   = (const float*)d_in[7];
  const float* bk   = (const float*)d_in[8];
  const float* Wv   = (const float*)d_in[9];
  const float* bv   = (const float*)d_in[10];
  const float* qsc  = (const float*)d_in[11];
  const float* ksc  = (const float*)d_in[12];
  const float* hsc  = (const float*)d_in[13];
  const float* Wo   = (const float*)d_in[14];
  const float* bo   = (const float*)d_in[15];
  float* out = (float*)d_out;

  char* ws = (char*)d_ws;
  const size_t MB = 1024ull * 1024ull;
  __bf16* xq   = (__bf16*)(ws + 0 * MB);     // 16 MB  q  activations bf16
  __bf16* xkv  = (__bf16*)(ws + 16 * MB);    // 16 MB  kv activations bf16
  __bf16* WqT  = (__bf16*)(ws + 32 * MB);    //  2 MB
  __bf16* WkT  = (__bf16*)(ws + 34 * MB);    //  2 MB
  __bf16* WvT  = (__bf16*)(ws + 36 * MB);    //  2 MB
  __bf16* WoT  = (__bf16*)(ws + 38 * MB);    //  2 MB
  __bf16* qhb  = (__bf16*)(ws + 40 * MB);    // 16 MB  q heads (post norm+rope)
  __bf16* khb  = (__bf16*)(ws + 56 * MB);    // 16 MB  k heads (post norm+rope)
  __bf16* vTb  = (__bf16*)(ws + 72 * MB);    // 16 MB  v transposed [b][h][d][T]
  __bf16* atnb = (__bf16*)(ws + 88 * MB);    // 16 MB  attention output

  const int nElem = M_ * D_;
  f32_to_bf16_kernel<<<nElem / 256, 256, 0, stream>>>(q,  xq,  nElem);
  f32_to_bf16_kernel<<<nElem / 256, 256, 0, stream>>>(kv, xkv, nElem);

  dim3 tb(16, 16), tg(D_ / 16, D_ / 16);
  transpose_f32_to_bf16<<<tg, tb, 0, stream>>>(Wq, WqT, D_, D_);
  transpose_f32_to_bf16<<<tg, tb, 0, stream>>>(Wk, WkT, D_, D_);
  transpose_f32_to_bf16<<<tg, tb, 0, stream>>>(Wv, WvT, D_, D_);
  transpose_f32_to_bf16<<<tg, tb, 0, stream>>>(Wo, WoT, D_, D_);

  dim3 gg(H_, M_ / 32);   // (16, 256) waves: head x 32-row tile
  gemm_qkv_proj<<<gg, 32, 0, stream>>>(xq,  WqT, bq, qsc, qpos, qhb, nullptr, 0);
  gemm_qkv_proj<<<gg, 32, 0, stream>>>(xkv, WkT, bk, ksc, kpos, khb, nullptr, 0);
  gemm_qkv_proj<<<gg, 32, 0, stream>>>(xkv, WvT, bv, nullptr, nullptr, nullptr, vTb, 1);

  dim3 ga(T_ / 32, H_, B_);
  flash_attn_fwd<<<ga, 32, 0, stream>>>(qhb, khb, vTb, hsc, atnb);

  gemm_out_proj<<<gg, 32, 0, stream>>>(atnb, WoT, bo, out);
}
